// Generator_37538014167195
// MI455X (gfx1250) — compile-verified, matrix operand order checked by reading, not run on hardware
//
#include <hip/hip_runtime.h>
#include <hip/hip_bf16.h>

// ---------------------------------------------------------------------------
// StyleGAN2 generator forward for MI455X (gfx1250).
// Heavy 3x3 modulated convs run as implicit GEMM on V_WMMA_F32_16X16X32_F16
// (wave32). Register macro-tile: MT x NT wmma tiles (32 Cout x 32 pixels for
// the main layers) so each A/B fragment feeds 2 wmma -> 2 b128 loads per wmma.
// Activations are zero-padded NHWC f16 (1-pixel halo): branch-free inner loop.
// Upsample convs use analytically-folded (conv_transpose + 4x4 blur) per-parity
// effective 3x3 kernels so they reuse the same WMMA kernel.
// ---------------------------------------------------------------------------

typedef _Float16 h8   __attribute__((ext_vector_type(8)));
typedef _Float16 v16h __attribute__((ext_vector_type(16)));
typedef float    v8f  __attribute__((ext_vector_type(8)));

#define SQRT2F 1.41421356237309515f
#define INV_SQRT_512 0.044194173824159216f

__device__ __forceinline__ float kkf(int i) {
    // blur taps [1,3,3,1], zero outside
    return (i == 0 || i == 3) ? 1.f : ((i == 1 || i == 2) ? 3.f : 0.f);
}

// ---------------------------------------------------------------------------
// PixelNorm + 8-layer style MLP. One block per batch element.
// ---------------------------------------------------------------------------
__global__ void __launch_bounds__(512)
mlp_kernel(const float* __restrict__ z, const float* __restrict__ mlpw,
           const float* __restrict__ mlpb, float* __restrict__ styles) {
    __shared__ float s[512];
    __shared__ float red[512];
    const int b = blockIdx.x, t = threadIdx.x;
    float v = z[b * 512 + t];
    red[t] = v * v;
    __syncthreads();
    for (int o = 256; o > 0; o >>= 1) { if (t < o) red[t] += red[t + o]; __syncthreads(); }
    v *= rsqrtf(red[0] * (1.f / 512.f) + 1e-8f);
    s[t] = v;
    __syncthreads();
    const float scale = 0.01f * INV_SQRT_512;     // lr_mul / sqrt(in)
    for (int L = 0; L < 8; ++L) {
        const float* wr = mlpw + ((size_t)L * 512 + t) * 512;
        float a = 0.f;
        for (int j = 0; j < 512; ++j) a += wr[j] * s[j];
        a = a * scale + mlpb[L * 512 + t] * 0.01f;
        a = (a < 0.f ? 0.2f * a : a) * SQRT2F;
        __syncthreads();
        s[t] = a;
        __syncthreads();
    }
    styles[b * 512 + t] = s[t];
}

// ---------------------------------------------------------------------------
// Per-layer style modulation: smod[b][ci] = style[b].mod_w[ci]/sqrt(512)+mod_b
// ---------------------------------------------------------------------------
__global__ void __launch_bounds__(256)
style_mod_kernel(const float* __restrict__ styles, const float* __restrict__ modw,
                 const float* __restrict__ modb, float* __restrict__ smod, int Cin) {
    int idx = blockIdx.x * 256 + threadIdx.x;
    if (idx >= 4 * Cin) return;
    int b = idx / Cin, ci = idx % Cin;
    const float* st = styles + b * 512;
    const float* wr = modw + (size_t)ci * 512;
    float a = 0.f;
    for (int j = 0; j < 512; ++j) a += st[j] * wr[j];
    smod[idx] = a * INV_SQRT_512 + modb[ci];
}

// ---------------------------------------------------------------------------
// Modulate + demodulate weights, emit f16 in GEMM-friendly [b][tap][co][ci].
// up==1: fold conv_transpose(stride2) + 4x4 blur into per-parity effective 3x3.
// ---------------------------------------------------------------------------
__global__ void __launch_bounds__(256)
prep_weights(const float* __restrict__ weight,  // [Cout][Cin][3][3]
             const float* __restrict__ smod,    // [B][Cin]
             _Float16* __restrict__ wout,       // [B][9][Cout][Cin]
             int Cin, int Cout, int up, int py, int px) {
    __shared__ float red[256];
    const int co = blockIdx.x, b = blockIdx.y, t = threadIdx.x;
    const float scale = rsqrtf((float)(Cin * 9));
    const float* wrow = weight + (size_t)co * Cin * 9;
    const float* sb   = smod + (size_t)b * Cin;
    float ss = 0.f;
    for (int ci = t; ci < Cin; ci += 256) {
        float s = sb[ci];
        const float* w9 = wrow + ci * 9;
        for (int k = 0; k < 9; ++k) { float wv = scale * w9[k] * s; ss += wv * wv; }
    }
    red[t] = ss;
    __syncthreads();
    for (int o = 128; o > 0; o >>= 1) { if (t < o) red[t] += red[t + o]; __syncthreads(); }
    const float d = rsqrtf(red[0] + 1e-8f);
    _Float16* wo = wout + (size_t)b * 9 * Cout * Cin;
    for (int ci = t; ci < Cin; ci += 256) {
        float s = sb[ci];
        const float* w9 = wrow + ci * 9;
        float wm[9];
        for (int k = 0; k < 9; ++k) wm[k] = scale * w9[k] * s * d;
        if (!up) {
            for (int k = 0; k < 9; ++k)
                wo[((size_t)k * Cout + co) * Cin + ci] = (_Float16)wm[k];
        } else {
            for (int di = 0; di < 3; ++di)
                for (int dj = 0; dj < 3; ++dj) {
                    float e = 0.f;
                    for (int u = 0; u < 3; ++u) {
                        float cy = kkf(2 * di - 1 - py + u);
                        if (cy == 0.f) continue;
                        for (int v = 0; v < 3; ++v) {
                            float cx = kkf(2 * dj - 1 - px + v);
                            e += cy * cx * wm[u * 3 + v];
                        }
                    }
                    wo[((size_t)(di * 3 + dj) * Cout + co) * Cin + ci] =
                        (_Float16)(e * (1.f / 16.f));
                }
        }
    }
}

// ---------------------------------------------------------------------------
// Implicit-GEMM 3x3 conv via V_WMMA_F32_16X16X32_F16. One wave per block.
// MT co-tiles x NT pixel-tiles of 16: each fragment feeds multiple wmma.
// Activations zero-padded NHWC: inner loop is branch-free.
// up==1: H,Wd are low-res input dims; launch covers output parity (py,px).
// ---------------------------------------------------------------------------
template <int MT, int NT>
__global__ void __launch_bounds__(32)
conv3x3_wmma(const _Float16* __restrict__ xin,   // [B][(H+2)*(W+2)][Cin]
             const _Float16* __restrict__ wgt,   // [B][9][Cout][Cin]
             const float* __restrict__ noise,    // [Hout*Wout]
             const float* __restrict__ actb,     // [Cout]
             const float* __restrict__ noisew,   // scalar
             _Float16* __restrict__ xout,        // [B][(Hout+2)*(Wout+2)][Cout]
             int H, int Wd, int Cin, int Cout, int up, int py, int px) {
    const int lane = threadIdx.x;
    const int m    = lane & 15;
    const int hi   = lane >> 4;
    const int kb0  = hi << 3;
    const int b    = blockIdx.z;
    const int co0  = blockIdx.y * (16 * MT);
    const int PW   = Wd + 2;

    const _Float16* xb = xin + (size_t)b * (H + 2) * PW * Cin;
    const _Float16* wbase[MT];
#pragma unroll
    for (int mt = 0; mt < MT; ++mt)
        wbase[mt] = wgt + ((size_t)b * 9 * Cout + (co0 + 16 * mt + m)) * Cin;

    int Yt[NT], Xt[NT];
    const _Float16* xt[NT];
#pragma unroll
    for (int t = 0; t < NT; ++t) {
        const int p = blockIdx.x * (16 * NT) + 16 * t + m;
        Yt[t] = p / Wd;
        Xt[t] = p % Wd;
        xt[t] = xb + (size_t)(Yt[t] * PW + Xt[t]) * Cin;   // tap(0,0) base in padded grid
    }

    const v8f zf = {0.f, 0.f, 0.f, 0.f, 0.f, 0.f, 0.f, 0.f};
    v8f acc[MT][NT];
#pragma unroll
    for (int mt = 0; mt < MT; ++mt)
#pragma unroll
        for (int t = 0; t < NT; ++t) acc[mt][t] = zf;

    for (int tap = 0; tap < 9; ++tap) {
        const size_t wo = (size_t)tap * Cout * Cin;
        const size_t xo = (size_t)((tap / 3) * PW + (tap % 3)) * Cin;
        for (int c0 = 0; c0 < Cin; c0 += 32) {
            v16h A[MT], B[NT];
#pragma unroll
            for (int mt = 0; mt < MT; ++mt) {
                h8 a0 = *(const h8*)(wbase[mt] + wo + c0 + kb0);
                h8 a1 = *(const h8*)(wbase[mt] + wo + c0 + kb0 + 16);
                A[mt] = __builtin_shufflevector(a0, a1, 0,1,2,3,4,5,6,7,8,9,10,11,12,13,14,15);
            }
#pragma unroll
            for (int t = 0; t < NT; ++t) {
                h8 b0 = *(const h8*)(xt[t] + xo + c0 + kb0);
                h8 b1 = *(const h8*)(xt[t] + xo + c0 + kb0 + 16);
                B[t] = __builtin_shufflevector(b0, b1, 0,1,2,3,4,5,6,7,8,9,10,11,12,13,14,15);
            }
#pragma unroll
            for (int mt = 0; mt < MT; ++mt)
#pragma unroll
                for (int t = 0; t < NT; ++t)
                    acc[mt][t] = __builtin_amdgcn_wmma_f32_16x16x32_f16(
                        /*neg_a=*/false, A[mt], /*neg_b=*/false, B[t],
                        /*c_mod=*/(short)0, acc[mt][t],
                        /*reuse_a=*/false, /*reuse_b=*/false);
        }
    }

    const float nw   = noisew[0];
    const int   Wout = up ? (Wd << 1) : Wd;
    const int   Hout = up ? (H << 1) : H;
    const int   PWo  = Wout + 2;

#pragma unroll
    for (int t = 0; t < NT; ++t) {
        const int oy = up ? ((Yt[t] << 1) | py) : Yt[t];
        const int ox = up ? ((Xt[t] << 1) | px) : Xt[t];
        const float nz = nw * noise[oy * Wout + ox];
        _Float16* orow =
            xout + (((size_t)b * (Hout + 2) + (oy + 1)) * PWo + (ox + 1)) * Cout;
#pragma unroll
        for (int mt = 0; mt < MT; ++mt) {
            const int cob = co0 + 16 * mt + (hi << 3);
            h8 st;
#pragma unroll
            for (int r = 0; r < 8; ++r) {
                float v = acc[mt][t][r] + nz + actb[cob + r];
                v = (v < 0.f ? 0.2f * v : v) * SQRT2F;
                st[r] = (_Float16)v;
            }
            *(h8*)(orow + cob) = st;
        }
    }
}

// ---------------------------------------------------------------------------
// Zero the 1-pixel halo of a padded NHWC activation buffer.
// Grid.x enumerates border positions only (2*(W+2) + 2*H).
// ---------------------------------------------------------------------------
__global__ void __launch_bounds__(64)
zero_border_kernel(_Float16* __restrict__ buf, int H, int W, int C) {
    const int b  = blockIdx.y;
    const int PW = W + 2;
    int idx = blockIdx.x, y, x;
    if (idx < PW)            { y = 0;     x = idx; }
    else if (idx < 2 * PW)   { y = H + 1; x = idx - PW; }
    else { int r = idx - 2 * PW; y = 1 + (r >> 1); x = (r & 1) ? (W + 1) : 0; }
    _Float16* p = buf + (((size_t)b * (H + 2) + y) * PW + x) * C;
    const h8 hz = {0, 0, 0, 0, 0, 0, 0, 0};
    for (int c = threadIdx.x; c < (C >> 3); c += 64) ((h8*)p)[c] = hz;
}

// ---------------------------------------------------------------------------
// Modulated 1x1 toRGB (no demod) + bias + upsampled-blur skip accumulate.
// Skip layout: [B][3][H][H] f32 (matches reference output layout).
// ---------------------------------------------------------------------------
__global__ void __launch_bounds__(256)
to_rgb_kernel(const _Float16* __restrict__ act,   // padded NHWC [B][(H+2)^2][Cin]
              const float* __restrict__ smod,     // [B][Cin]
              const float* __restrict__ wrgb,     // [3][Cin]
              const float* __restrict__ bias,     // [3]
              const float* __restrict__ skipOld,  // [B][3][H/2][H/2] or null
              float* __restrict__ skipNew,        // [B][3][H][H]
              int H, int Cin) {
    __shared__ float wm[3 * 512];
    const int b = blockIdx.y, t = threadIdx.x;
    const float scale = rsqrtf((float)Cin);
    for (int i = t; i < 3 * Cin; i += 256)
        wm[i] = scale * wrgb[i] * smod[b * Cin + (i % Cin)];
    __syncthreads();
    const int HW = H * H;
    const int pid = blockIdx.x * 256 + t;
    if (pid >= HW) return;
    const int y = pid / H, x = pid % H;
    const int PH = H + 2;
    const _Float16* xr = act + (((size_t)b * PH + (y + 1)) * PH + (x + 1)) * Cin;
    float a0 = 0.f, a1 = 0.f, a2 = 0.f;
    for (int ci = 0; ci < Cin; ++ci) {
        float xv = (float)xr[ci];
        a0 += wm[ci] * xv;
        a1 += wm[Cin + ci] * xv;
        a2 += wm[2 * Cin + ci] * xv;
    }
    float o[3] = {a0 + bias[0], a1 + bias[1], a2 + bias[2]};
    if (skipOld) {
        const int Hs = H >> 1, Yb = y >> 1, Xb = x >> 1;
        float cy0, cy1, cx0, cx1;
        int iy0, iy1, ix0, ix1;
        if ((y & 1) == 0) { cy0 = 1.f; iy0 = Yb - 1; cy1 = 3.f; iy1 = Yb; }
        else              { cy0 = 3.f; iy0 = Yb;     cy1 = 1.f; iy1 = Yb + 1; }
        if ((x & 1) == 0) { cx0 = 1.f; ix0 = Xb - 1; cx1 = 3.f; ix1 = Xb; }
        else              { cx0 = 3.f; ix0 = Xb;     cx1 = 1.f; ix1 = Xb + 1; }
        for (int c = 0; c < 3; ++c) {
            const float* sp = skipOld + ((size_t)b * 3 + c) * Hs * Hs;
            auto tapv = [&](int iy, int ix) -> float {
                return (iy >= 0 && iy < Hs && ix >= 0 && ix < Hs) ? sp[iy * Hs + ix] : 0.f;
            };
            float s = cy0 * (cx0 * tapv(iy0, ix0) + cx1 * tapv(iy0, ix1)) +
                      cy1 * (cx0 * tapv(iy1, ix0) + cx1 * tapv(iy1, ix1));
            o[c] += s * (1.f / 16.f);
        }
    }
    for (int c = 0; c < 3; ++c)
        skipNew[((size_t)b * 3 + c) * HW + pid] = o[c];
}

// ---------------------------------------------------------------------------
// Broadcast const_input [512][4][4] f32 -> padded NHWC f16 [B][6*6][512]
// (interior filled, halo zeroed).
// ---------------------------------------------------------------------------
__global__ void __launch_bounds__(256)
init_const_kernel(const float* __restrict__ ci, _Float16* __restrict__ act) {
    const int b = blockIdx.y;
    const int idx = blockIdx.x * 256 + threadIdx.x;   // ppos*512 + c, ppos in 6x6
    if (idx >= 36 * 512) return;
    const int ppos = idx >> 9, c = idx & 511;
    const int y = ppos / 6, x = ppos % 6;
    const bool interior = (y >= 1 && y <= 4 && x >= 1 && x <= 4);
    act[(size_t)b * 36 * 512 + idx] =
        interior ? (_Float16)ci[c * 16 + (y - 1) * 4 + (x - 1)] : (_Float16)0.f;
}

// ---------------------------------------------------------------------------
// Host orchestration
// ---------------------------------------------------------------------------
extern "C" void kernel_launch(void* const* d_in, const int* in_sizes, int n_in,
                              void* d_out, int out_size, void* d_ws, size_t ws_size,
                              hipStream_t stream) {
    (void)in_sizes; (void)n_in; (void)out_size; (void)ws_size;
    const int CH[6] = {512, 512, 512, 512, 512, 256};

    const float* z      = (const float*)d_in[0];
    const float* mlp_w  = (const float*)d_in[1];
    const float* mlp_b  = (const float*)d_in[2];
    const float* cinput = (const float*)d_in[3];
    auto CP = [&](int L, int f) { return (const float*)d_in[4 + 5 * L + f]; };  // weight,mod_w,mod_b,noise_w,act_b
    auto RP = [&](int t, int f) { return (const float*)d_in[59 + 4 * t + f]; }; // weight,mod_w,mod_b,bias
    auto NZ = [&](int li)       { return (const float*)d_in[83 + li]; };

    // Workspace carve (256B aligned)
    char* wsp = (char*)d_ws;
    auto alloc = [&](size_t bytes) { void* p = wsp; wsp += (bytes + 255) & ~(size_t)255; return p; };
    float*     styles = (float*)alloc(4 * 512 * sizeof(float));
    float*     smod   = (float*)alloc(4 * 512 * sizeof(float));
    _Float16*  wbuf   = (_Float16*)alloc((size_t)4 * 9 * 512 * 512 * sizeof(_Float16));
    const size_t actHalves = (size_t)4 * 130 * 130 * 256;  // max padded pos*chan over layers
    _Float16*  actA   = (_Float16*)alloc(actHalves * sizeof(_Float16));
    _Float16*  actB   = (_Float16*)alloc(actHalves * sizeof(_Float16));
    float*     skipA  = (float*)alloc((size_t)4 * 3 * 128 * 128 * sizeof(float));
    float*     skipB  = (float*)alloc((size_t)4 * 3 * 128 * 128 * sizeof(float));

    mlp_kernel<<<4, 512, 0, stream>>>(z, mlp_w, mlp_b, styles);
    init_const_kernel<<<dim3(72, 4), 256, 0, stream>>>(cinput, actA);

    auto launch_conv = [&](int L, const _Float16* in, _Float16* out, int res,
                           int Cin, int Cout, int up, int nidx) {
        const int rout = up ? 2 * res : res;
        // halo of the output must be zero for the next layer's reads
        zero_border_kernel<<<dim3(2 * (rout + 2) + 2 * rout, 4), 64, 0, stream>>>(
            out, rout, rout, Cout);
        style_mod_kernel<<<(4 * Cin + 255) / 256, 256, 0, stream>>>(
            styles, CP(L, 1), CP(L, 2), smod, Cin);
        const int HW = res * res;
        const bool big = (HW % 32 == 0) && (Cout % 32 == 0);
        auto conv = [&](int py, int px) {
            if (big)
                conv3x3_wmma<2, 2><<<dim3(HW / 32, Cout / 32, 4), 32, 0, stream>>>(
                    in, wbuf, NZ(nidx), CP(L, 4), CP(L, 3), out, res, res, Cin, Cout, up, py, px);
            else
                conv3x3_wmma<1, 1><<<dim3(HW / 16, Cout / 16, 4), 32, 0, stream>>>(
                    in, wbuf, NZ(nidx), CP(L, 4), CP(L, 3), out, res, res, Cin, Cout, up, py, px);
        };
        if (!up) {
            prep_weights<<<dim3(Cout, 4), 256, 0, stream>>>(
                CP(L, 0), smod, wbuf, Cin, Cout, 0, 0, 0);
            conv(0, 0);
        } else {
            for (int py = 0; py < 2; ++py)
                for (int px = 0; px < 2; ++px) {
                    prep_weights<<<dim3(Cout, 4), 256, 0, stream>>>(
                        CP(L, 0), smod, wbuf, Cin, Cout, 1, py, px);
                    conv(py, px);
                }
        }
    };
    auto launch_rgb = [&](int ti, const _Float16* act, const float* skOld,
                          float* skNew, int res, int Cin) {
        style_mod_kernel<<<(4 * Cin + 255) / 256, 256, 0, stream>>>(
            styles, RP(ti, 1), RP(ti, 2), smod, Cin);
        const int HW = res * res;
        to_rgb_kernel<<<dim3((HW + 255) / 256, 4), 256, 0, stream>>>(
            act, smod, RP(ti, 0), RP(ti, 3), skOld, skNew, res, Cin);
    };

    // Layer 0 at res 4 + first toRGB
    launch_conv(0, actA, actB, 4, 512, 512, /*up=*/0, /*noise=*/0);
    launch_rgb(0, actB, nullptr, skipA, 4, 512);

    _Float16 *x = actB, *y = actA;
    float* sc = skipA;   // current skip
    int res = 4;
    for (int li = 0; li < 5; ++li) {
        const int Cin = CH[li], Cout = CH[li + 1];
        launch_conv(1 + 2 * li, x, y, res, Cin, Cout, /*up=*/1, 1 + 2 * li);   // res -> 2*res
        res <<= 1;
        launch_conv(2 + 2 * li, y, x, res, Cout, Cout, /*up=*/0, 2 + 2 * li);
        float* dst = (li == 4) ? (float*)d_out
                               : ((sc == skipA) ? skipB : skipA);
        launch_rgb(1 + li, x, sc, dst, res, Cout);
        sc = dst;
        { _Float16* t2 = x; x = y; y = t2; }  // ping-pong activations
    }
}
